// Block_18554258718990
// MI455X (gfx1250) — 8-bit matrix operand order repaired and checked by reading, compiled, not run on hardware
//
#include <hip/hip_runtime.h>
#include <hip/hip_bf16.h>
#include <stdint.h>

#define EPSF 1e-5f

typedef __attribute__((ext_vector_type(8))) int v8i;
typedef __attribute__((ext_vector_type(4))) int v4i;

static const int Dm   = 4096;
static const int Mtok = 8192;

#if __has_builtin(__builtin_amdgcn_global_load_async_to_lds_b128) && \
    __has_builtin(__builtin_amdgcn_s_wait_asynccnt)
#define HAVE_ASYNC_LDS 1
#else
#define HAVE_ASYNC_LDS 0
#endif

__global__ void absmean_partial_kernel(const float* __restrict__ W,
                                       float* __restrict__ partial, int n) {
    __shared__ float sm[256];
    float s = 0.0f;
    for (int i = blockIdx.x * blockDim.x + threadIdx.x; i < n;
         i += gridDim.x * blockDim.x)
        s += fabsf(W[i]);
    sm[threadIdx.x] = s;
    __syncthreads();
    for (int off = 128; off > 0; off >>= 1) {
        if ((int)threadIdx.x < off) sm[threadIdx.x] += sm[threadIdx.x + off];
        __syncthreads();
    }
    if (threadIdx.x == 0) partial[blockIdx.x] = sm[0];
}

__global__ void absmean_final_kernel(const float* __restrict__ partial,
                                     float* __restrict__ wdeq,
                                     int nparts, float inv_count) {
    __shared__ float sm[256];
    float s = 0.0f;
    for (int i = threadIdx.x; i < nparts; i += 256) s += partial[i];
    sm[threadIdx.x] = s;
    __syncthreads();
    for (int off = 128; off > 0; off >>= 1) {
        if ((int)threadIdx.x < off) sm[threadIdx.x] += sm[threadIdx.x + off];
        __syncthreads();
    }
    if (threadIdx.x == 0) {
        float mean = sm[0] * inv_count;
        wdeq[0] = fmaxf(mean, EPSF);
    }
}

__global__ void quant_w_kernel(const float* __restrict__ W,
                               signed char* __restrict__ W8,
                               const float* __restrict__ wdeq, int n4) {
    const float wsc = 1.0f / wdeq[0];
    const int stride = gridDim.x * blockDim.x;
    for (int i = blockIdx.x * blockDim.x + threadIdx.x; i < n4; i += stride) {
        float4 w = ((const float4*)W)[i];
        char4 q;
        q.x = (signed char)fminf(fmaxf(rintf(w.x * wsc), -1.0f), 1.0f);
        q.y = (signed char)fminf(fmaxf(rintf(w.y * wsc), -1.0f), 1.0f);
        q.z = (signed char)fminf(fmaxf(rintf(w.z * wsc), -1.0f), 1.0f);
        q.w = (signed char)fminf(fmaxf(rintf(w.w * wsc), -1.0f), 1.0f);
        ((char4*)W8)[i] = q;
    }
}

__global__ void quant_act_kernel(const float* __restrict__ x,
                                 signed char* __restrict__ A8,
                                 float* __restrict__ actdeq) {
    const int D = Dm;
    const int tid = threadIdx.x;
    const size_t tok = blockIdx.x;
    const float* xr = x + tok * (size_t)D;
    __shared__ float sm[256];
    __shared__ float s_rms, s_scale;

    float ss = 0.0f, am = 0.0f;
    for (int i = tid; i < D; i += 256) {
        float v = xr[i];
        ss += v * v;
        am = fmaxf(am, fabsf(v));
    }
    sm[tid] = ss;
    __syncthreads();
    for (int off = 128; off > 0; off >>= 1) {
        if (tid < off) sm[tid] += sm[tid + off];
        __syncthreads();
    }
    if (tid == 0) s_rms = rsqrtf(sm[0] / (float)D + EPSF);
    __syncthreads();
    const float rms = s_rms;

    sm[tid] = am;
    __syncthreads();
    for (int off = 128; off > 0; off >>= 1) {
        if (tid < off) sm[tid] = fmaxf(sm[tid], sm[tid + off]);
        __syncthreads();
    }
    if (tid == 0) {
        float amax = fmaxf(sm[0] * rms, EPSF);
        s_scale = 127.0f / amax;
        actdeq[tok] = amax / 127.0f;
    }
    __syncthreads();
    const float sc = s_scale * rms;
    signed char* arow = A8 + tok * (size_t)D;
    for (int i = tid; i < D; i += 256) {
        float q = rintf(xr[i] * sc);
        q = fminf(fmaxf(q, -128.0f), 127.0f);
        arow[i] = (signed char)q;
    }
}

#define ROWP 80

__global__ void __launch_bounds__(256) bitnet_wmma_gemm(
    const signed char* __restrict__ A8,
    const signed char* __restrict__ W8,
    const float* __restrict__ actdeq,
    const float* __restrict__ wdeqp,
    const float* __restrict__ bias,
    const float* __restrict__ xres,
    float* __restrict__ out)
{
    const int K = Dm;
    const int N = Dm;
    __shared__ __align__(16) signed char lA[3][128 * ROWP];
    __shared__ __align__(16) signed char lB[3][128 * ROWP];

    const int t     = threadIdx.x;
    const int lane  = t & 31;
    const int wave  = t >> 5;
    const int l16   = lane & 15;
    const int half  = lane >> 4;
    const int waveN = wave & 3;
    const int waveM = wave >> 2;

    const int n0 = blockIdx.x * 128;
    const int m0 = blockIdx.y * 128;

    const int sRow = t >> 2;
    const int sCol = (t & 3) * 16;

    auto stage = [&](int buf, int k0) {
#pragma unroll
        for (int p = 0; p < 2; ++p) {
            const int r = sRow + 64 * p;
            const signed char* ga = &A8[(size_t)(m0 + r) * K + k0 + sCol];
            const signed char* gb = &W8[(size_t)(n0 + r) * K + k0 + sCol];
            signed char* la = &lA[buf][r * ROWP + sCol];
            signed char* lb = &lB[buf][r * ROWP + sCol];
#if HAVE_ASYNC_LDS
            __builtin_amdgcn_global_load_async_to_lds_b128(
                (v4i*)ga, (v4i*)la, 0, 0);
            __builtin_amdgcn_global_load_async_to_lds_b128(
                (v4i*)gb, (v4i*)lb, 0, 0);
#else
            *(int4*)la = *(const int4*)ga;
            *(int4*)lb = *(const int4*)gb;
            __builtin_prefetch(ga + 64, 0, 0);
            __builtin_prefetch(gb + 64, 0, 0);
#endif
        }
    };

    v8i acc[4][2] = {};

    auto compute = [&](int buf) {
        v8i bf[2];
#pragma unroll
        for (int tn = 0; tn < 2; ++tn) {
            const signed char* bp =
                &lB[buf][(waveN * 32 + tn * 16 + l16) * ROWP + half * 16];
            union { v8i v; int4 q[2]; } ub;
            ub.q[0] = *(const int4*)(bp);
            ub.q[1] = *(const int4*)(bp + 32);
            bf[tn] = ub.v;
        }
#pragma unroll
        for (int tm = 0; tm < 4; ++tm) {
            const uint64_t* ap = (const uint64_t*)
                &lA[buf][(waveM * 64 + tm * 16 + l16) * ROWP + half * 16];
            union { v8i v; uint64_t u[4]; } ua;
            ua.u[0] = ap[0];
            ua.u[1] = ap[1];
            ua.u[2] = ap[4];
            ua.u[3] = ap[5];
#pragma unroll
            for (int tn = 0; tn < 2; ++tn) {
                acc[tm][tn] = __builtin_amdgcn_wmma_i32_16x16x64_iu8(
                     true, ua.v,  true, bf[tn],
                    acc[tm][tn],  false,  false);
            }
        }
    };

    const int nt = K / 64;
    stage(0, 0);
    stage(1, 64);

    int bufC = 0;
    int bufS = 2;
    for (int i = 0; i < nt - 2; ++i) {
#if HAVE_ASYNC_LDS
        __builtin_amdgcn_s_wait_asynccnt(4);
#endif
        __syncthreads();
        stage(bufS, (i + 2) * 64);
        compute(bufC);
        bufC = (bufC == 2) ? 0 : bufC + 1;
        bufS = (bufS == 2) ? 0 : bufS + 1;
    }
#if HAVE_ASYNC_LDS
    __builtin_amdgcn_s_wait_asynccnt(4);
#endif
    __syncthreads();
    compute(bufC);
    bufC = (bufC == 2) ? 0 : bufC + 1;
#if HAVE_ASYNC_LDS
    __builtin_amdgcn_s_wait_asynccnt(0);
#endif
    __syncthreads();
    compute(bufC);

    const float wd = wdeqp[0];
#pragma unroll
    for (int tm = 0; tm < 4; ++tm) {
        const int gm = m0 + waveM * 64 + tm * 16 + half * 8;
#pragma unroll
        for (int tn = 0; tn < 2; ++tn) {
            const int gn = n0 + waveN * 32 + tn * 16 + l16;
            const float bcol = bias[gn];
#pragma unroll
            for (int r = 0; r < 8; ++r) {
                const int m = gm + r;
                float y = (float)acc[tm][tn][r] * (actdeq[m] * wd) + bcol;
                float g = 0.5f * y * (1.0f + erff(y * 0.70710678118654752f));
                const size_t idx = (size_t)m * N + gn;
                out[idx] = xres[idx] + g;
            }
        }
    }
}

extern "C" void kernel_launch(void* const* d_in, const int* in_sizes, int n_in,
                              void* d_out, int out_size, void* d_ws, size_t ws_size,
                              hipStream_t stream) {
    const float* x = (const float*)d_in[0];
    const float* W = (const float*)d_in[1];
    const float* b = (const float*)d_in[2];
    float* out = (float*)d_out;

    char* ws = (char*)d_ws;
    signed char* A8  = (signed char*)(ws);
    signed char* W8  = (signed char*)(ws + 33554432);
    float* actdeq    = (float*)(ws + 33554432 + 16777216);
    float* partial   = (float*)(ws + 33554432 + 16777216 + 32768);
    float* wdeq      = (float*)(ws + 33554432 + 16777216 + 32768 + 4096);

    absmean_partial_kernel<<<1024, 256, 0, stream>>>(W, partial, Dm * Dm);
    absmean_final_kernel<<<1, 256, 0, stream>>>(partial, wdeq, 1024,
                                                1.0f / (float)(Dm * Dm));
    quant_w_kernel<<<4096, 256, 0, stream>>>(W, W8, wdeq, (Dm * Dm) / 4);
    quant_act_kernel<<<Mtok, 256, 0, stream>>>(x, A8, actdeq);

    dim3 grid(Dm / 128, Mtok / 128);
    bitnet_wmma_gemm<<<grid, 256, 0, stream>>>(A8, W8, actdeq, wdeq, b, x, out);
}
